// SoftmaxAttention_51436528337318
// MI455X (gfx1250) — matrix operand order repaired and checked by reading, not compiled, not run on hardware
//
#include <hip/hip_runtime.h>
#include <hip/hip_bf16.h>

// ---------------------------------------------------------------------------
// SoftmaxAttention for MI455X (gfx1250): bf16 WMMA GEMMs + flash attention,
// with double-buffered async global->LDS staging.
// ---------------------------------------------------------------------------

typedef __bf16 bf16_t;
typedef __attribute__((ext_vector_type(16))) __bf16 v16bf;
typedef __attribute__((ext_vector_type(8)))  float  v8f;
typedef int b128_t __attribute__((vector_size(16)));   // matches builtin param

constexpr int BATCH = 2;
constexpr int SEQ   = 2048;
constexpr int DIM   = 2048;
constexpr int NH    = 16;
constexpr int HDIM  = 128;
constexpr int BN    = BATCH * SEQ;   // 4096 rows

#if __has_builtin(__builtin_amdgcn_global_load_async_to_lds_b128)
#define USE_ASYNC_LDS 1
#else
#define USE_ASYNC_LDS 0
#endif

// Copy 16 bytes global -> LDS. Async (ASYNCcnt-tracked, no VGPR data) when the
// toolchain exposes the gfx1250 builtin; plain load+store otherwise.
static __device__ __forceinline__ void copy16_to_lds(const bf16_t* __restrict__ g,
                                                     bf16_t* __restrict__ l) {
#if USE_ASYNC_LDS
  __builtin_amdgcn_global_load_async_to_lds_b128(
      (__attribute__((address_space(1))) b128_t*)g,
      (__attribute__((address_space(3))) b128_t*)l, 0, 0);
#else
  *reinterpret_cast<uint4*>(l) = *reinterpret_cast<const uint4*>(g);
#endif
}

static __device__ __forceinline__ void async_wait0() {
#if USE_ASYNC_LDS
#if __has_builtin(__builtin_amdgcn_s_wait_asynccnt)
  __builtin_amdgcn_s_wait_asynccnt(0);
#else
  asm volatile("s_wait_asynccnt 0" ::: "memory");
#endif
#endif
}

static __device__ __forceinline__ v8f wmma_bf16(v16bf a, v16bf b, v8f c) {
  // (neg_a, A, neg_b, B, c_mod, C, reuse_a, reuse_b)
  return __builtin_amdgcn_wmma_f32_16x16x32_bf16(false, a, false, b, (short)0, c,
                                                 false, false);
}

union Frag { v16bf v; unsigned u[8]; };

// ---------------------------------------------------------------------------
// fp32 -> bf16 elementwise convert
// ---------------------------------------------------------------------------
__global__ __launch_bounds__(256) void cvt_f32_bf16(const float* __restrict__ x,
                                                    bf16_t* __restrict__ y,
                                                    size_t n) {
  size_t i = (size_t)blockIdx.x * 256 + threadIdx.x;
  if (i < n) y[i] = (bf16_t)x[i];
}

// ---------------------------------------------------------------------------
// Tiled bf16 GEMM: C(M,N) fp32 = A(M,K) * B(K,N), row-major.
// 256 threads = 8 waves; block tile 128x128; wave tile 32x64; K-step 32.
// Double-buffered LDS staging: A tile async, B tile load-then-scatter.
// ---------------------------------------------------------------------------
__global__ __launch_bounds__(256) void gemm_bf16(const bf16_t* __restrict__ A,
                                                 const bf16_t* __restrict__ Bm,
                                                 float* __restrict__ C,
                                                 int M, int N, int K) {
  __shared__ alignas(16) bf16_t As[2][128][48];  // [buf][m][k]
  __shared__ alignas(16) bf16_t Bs[2][128][48];  // [buf][n][k]

  const int tid  = threadIdx.x;
  const int wid  = tid >> 5;
  const int lane = tid & 31;
  const int hi   = lane >> 4;
  const int lm   = lane & 15;
  const int waveM = wid >> 1;   // 0..3 -> 32-row slice
  const int waveN = wid & 1;    // 0..1 -> 64-col slice
  const int m0 = blockIdx.y * 128;
  const int n0 = blockIdx.x * 128;

  v8f acc[2][4];
#pragma unroll
  for (int i = 0; i < 2; ++i)
#pragma unroll
    for (int t = 0; t < 4; ++t) acc[i][t] = (v8f)0.0f;

  const int arow = tid >> 2, ac8 = tid & 3;    // A: 64 rows/pass, 8 elems
  const int bk   = tid >> 4, bc8 = tid & 15;   // B: 16 k-rows/pass, 8 elems

  auto stageA = [&](int kk, int buf) {
#pragma unroll
    for (int p = 0; p < 2; ++p) {
      int r = arow + p * 64;
      copy16_to_lds(A + (size_t)(m0 + r) * K + kk + ac8 * 8, &As[buf][r][ac8 * 8]);
    }
  };
  auto stageB = [&](int kk, int buf) {  // transpose [k][n] -> [n][k]
    union { uint4 q4; bf16_t h[8]; } t0, t1;   // both loads in flight first
    t0.q4 = *reinterpret_cast<const uint4*>(Bm + (size_t)(kk + bk) * N + n0 + bc8 * 8);
    t1.q4 = *reinterpret_cast<const uint4*>(Bm + (size_t)(kk + bk + 16) * N + n0 + bc8 * 8);
#pragma unroll
    for (int i = 0; i < 8; ++i) Bs[buf][bc8 * 8 + i][bk] = t0.h[i];
#pragma unroll
    for (int i = 0; i < 8; ++i) Bs[buf][bc8 * 8 + i][bk + 16] = t1.h[i];
  };

  stageA(0, 0);
  stageB(0, 0);
  async_wait0();
  __syncthreads();

  const int nT = K / 32;
  for (int t = 0; t < nT; ++t) {
    const int cur = t & 1;
    if (t + 1 < nT) {               // prefetch next tile into the other buffer
      stageA((t + 1) * 32, cur ^ 1);
      stageB((t + 1) * 32, cur ^ 1);
    }

    Frag a[2], b[4];
#pragma unroll
    for (int i = 0; i < 2; ++i) {
      const unsigned* src =
          reinterpret_cast<const unsigned*>(&As[cur][waveM * 32 + i * 16 + lm][0]);
#pragma unroll
      for (int j = 0; j < 8; ++j) {
        int k = 2 * j + hi * 8 + ((j >= 4) ? 8 : 0);   // 16-bit A layout
        a[i].u[j] = src[k >> 1];
      }
    }
#pragma unroll
    for (int t4 = 0; t4 < 4; ++t4) {
      const unsigned* src =
          reinterpret_cast<const unsigned*>(&Bs[cur][waveN * 64 + t4 * 16 + lm][0]);
#pragma unroll
      for (int j = 0; j < 8; ++j) {
        int k = 2 * j + hi * 8 + ((j >= 4) ? 8 : 0);   // 16-bit B layout
        b[t4].u[j] = src[k >> 1];
      }
    }
#pragma unroll
    for (int i = 0; i < 2; ++i)
#pragma unroll
      for (int t4 = 0; t4 < 4; ++t4) acc[i][t4] = wmma_bf16(a[i].v, b[t4].v, acc[i][t4]);

    async_wait0();
    __syncthreads();
  }

  // ---- store C (fp32): one base pointer, immediate offsets ----
  float* cp = C + (size_t)(m0 + waveM * 32 + hi * 8) * N + (n0 + waveN * 64 + lm);
#pragma unroll
  for (int i = 0; i < 2; ++i)
#pragma unroll
    for (int t4 = 0; t4 < 4; ++t4)
#pragma unroll
      for (int r = 0; r < 8; ++r)
        cp[(size_t)(i * 16 + r) * N + t4 * 16] = acc[i][t4][r];
}

// ---------------------------------------------------------------------------
// Fused RMSNorm + RoPE, one wave per (token, head) row of 128 elements.
// Input X: (b, n, h*d) fp32; output: (b, h, n, d) bf16.
// ---------------------------------------------------------------------------
__global__ __launch_bounds__(256) void rmsnorm_rope(const float* __restrict__ X,
                                                    const float* __restrict__ w,
                                                    const float* __restrict__ cosp,
                                                    const float* __restrict__ sinp,
                                                    bf16_t* __restrict__ out) {
  int row  = blockIdx.x * 8 + (threadIdx.x >> 5);  // over BN*NH rows
  int lane = threadIdx.x & 31;
  int h    = row % NH;
  int tok  = row / NH;            // 0..BN-1
  int n    = tok % SEQ;
  int b    = tok / SEQ;

  const float* x = X + (size_t)tok * (NH * HDIM) + h * HDIM;
  float v0 = x[lane], v1 = x[lane + 32], v2 = x[lane + 64], v3 = x[lane + 96];
  float ss = v0 * v0 + v1 * v1 + v2 * v2 + v3 * v3;
#pragma unroll
  for (int m = 1; m < 32; m <<= 1) ss += __shfl_xor(ss, m);
  float inv = rsqrtf(ss * (1.0f / 128.0f) + 1e-6f);
  float y0 = v0 * inv * w[lane],      y1 = v1 * inv * w[lane + 32];
  float y2 = v2 * inv * w[lane + 64], y3 = v3 * inv * w[lane + 96];

  const float* cs = cosp + (size_t)n * HDIM;
  const float* sn = sinp + (size_t)n * HDIM;
  // rotate_half: d<64 -> pair +64 negated; d>=64 -> pair -64
  float o0 = y0 * cs[lane]      - y2 * sn[lane];
  float o1 = y1 * cs[lane + 32] - y3 * sn[lane + 32];
  float o2 = y2 * cs[lane + 64] + y0 * sn[lane + 64];
  float o3 = y3 * cs[lane + 96] + y1 * sn[lane + 96];

  bf16_t* op = out + ((size_t)(b * NH + h) * SEQ + n) * HDIM;
  op[lane]      = (bf16_t)o0;
  op[lane + 32] = (bf16_t)o1;
  op[lane + 64] = (bf16_t)o2;
  op[lane + 96] = (bf16_t)o3;
}

// ---------------------------------------------------------------------------
// V: (b,n,h,d) fp32 -> (b,h,n,d) bf16
// ---------------------------------------------------------------------------
__global__ __launch_bounds__(256) void v_cvt(const float* __restrict__ X,
                                             bf16_t* __restrict__ out) {
  size_t i = (size_t)blockIdx.x * 256 + threadIdx.x;  // over BN*DIM
  int d = (int)(i % HDIM);
  size_t t = i / HDIM;
  int h = (int)(t % NH);
  size_t tok = t / NH;
  int n = (int)(tok % SEQ);
  int b = (int)(tok / SEQ);
  out[((size_t)(b * NH + h) * SEQ + n) * HDIM + d] = (bf16_t)X[i];
}

// ---------------------------------------------------------------------------
// Flash attention. Grid: x = q-tile (SEQ/128), y = b*h (32).
// Block: 8 waves; each wave owns 16 q rows; K/V streamed in 32-key chunks,
// double-buffered in LDS (K async, V load-then-scatter transpose).
// q,k,v: (b,h,n,d) bf16.  ctx out: (b,n,h,d) bf16.
// ---------------------------------------------------------------------------
__global__ __launch_bounds__(256) void flash_attn(const bf16_t* __restrict__ q,
                                                  const bf16_t* __restrict__ k,
                                                  const bf16_t* __restrict__ v,
                                                  bf16_t* __restrict__ ctx) {
  __shared__ alignas(16) bf16_t Ks[2][32][144];   // [buf][key][d]
  __shared__ alignas(16) bf16_t Vs[2][128][48];   // [buf][d][key]
  __shared__ alignas(4)  bf16_t Ps[8][16][40];    // per-wave P staging

  const int tid  = threadIdx.x;
  const int wid  = tid >> 5;
  const int lane = tid & 31;
  const int hi   = lane >> 4;
  const int lm   = lane & 15;

  const int bh = blockIdx.y;
  const int b  = bh / NH;
  const int h  = bh % NH;
  const bf16_t* Qb = q + (size_t)bh * SEQ * HDIM;
  const bf16_t* Kb = k + (size_t)bh * SEQ * HDIM;
  const bf16_t* Vb = v + (size_t)bh * SEQ * HDIM;
  const int q0 = blockIdx.x * 128 + wid * 16;

  // Q fragments: 16 rows x 128 d, 4 K-chunks of 32, held in registers.
  Frag qa[4];
#pragma unroll
  for (int c = 0; c < 4; ++c) {
    const unsigned* src =
        reinterpret_cast<const unsigned*>(Qb + (size_t)(q0 + lm) * HDIM + c * 32);
#pragma unroll
    for (int j = 0; j < 8; ++j) {
      int kk = 2 * j + hi * 8 + ((j >= 4) ? 8 : 0);
      qa[c].u[j] = src[kk >> 1];
    }
  }

  v8f o[8];
#pragma unroll
  for (int j = 0; j < 8; ++j) o[j] = (v8f)0.0f;
  float mrow[8], lrow[8];
#pragma unroll
  for (int r = 0; r < 8; ++r) { mrow[r] = -1e30f; lrow[r] = 0.0f; }

  const int kr = tid >> 4, c8 = tid & 15;     // cooperative-load coordinates
  const float scale = 0.08838834764831845f;   // 1/sqrt(128)

  auto stageK = [&](int c0, int buf) {
#pragma unroll
    for (int p = 0; p < 2; ++p)
      copy16_to_lds(Kb + (size_t)(c0 + kr + p * 16) * HDIM + c8 * 8,
                    &Ks[buf][kr + p * 16][c8 * 8]);
  };
  auto stageV = [&](int c0, int buf) {        // transpose [key][d] -> [d][key]
    union { uint4 q4; bf16_t hh[8]; } t0, t1; // both loads in flight first
    t0.q4 = *reinterpret_cast<const uint4*>(
        Vb + (size_t)(c0 + kr) * HDIM + c8 * 8);
    t1.q4 = *reinterpret_cast<const uint4*>(
        Vb + (size_t)(c0 + kr + 16) * HDIM + c8 * 8);
#pragma unroll
    for (int i = 0; i < 8; ++i) Vs[buf][c8 * 8 + i][kr] = t0.hh[i];
#pragma unroll
    for (int i = 0; i < 8; ++i) Vs[buf][c8 * 8 + i][kr + 16] = t1.hh[i];
  };

  stageK(0, 0);
  stageV(0, 0);
  async_wait0();
  __syncthreads();

  for (int c0 = 0; c0 < SEQ; c0 += 32) {
    const int cur = (c0 >> 5) & 1;
    if (c0 + 32 < SEQ) {            // prefetch next K/V chunk
      stageK(c0 + 32, cur ^ 1);
      stageV(c0 + 32, cur ^ 1);
    }

    // ---- S = Q * K^T  (two 16-key tiles, K-dim 128 in 4 chunks) ----
    v8f s[2];
#pragma unroll
    for (int t = 0; t < 2; ++t) {
      s[t] = (v8f)0.0f;
#pragma unroll
      for (int c = 0; c < 4; ++c) {
        Frag kb;
        const unsigned* src =
            reinterpret_cast<const unsigned*>(&Ks[cur][t * 16 + lm][c * 32]);
#pragma unroll
        for (int j = 0; j < 8; ++j) {
          int kk = 2 * j + hi * 8 + ((j >= 4) ? 8 : 0);
          kb.u[j] = src[kk >> 1];
        }
        s[t] = wmma_bf16(qa[c].v, kb.v, s[t]);
      }
    }

    // ---- online softmax update ----
    float fac[8];
#pragma unroll
    for (int r = 0; r < 8; ++r) {
      float s0 = s[0][r] * scale, s1 = s[1][r] * scale;
      float mx = fmaxf(s0, s1);
#pragma unroll
      for (int m = 1; m < 16; m <<= 1) mx = fmaxf(mx, __shfl_xor(mx, m));
      float mn = fmaxf(mrow[r], mx);
      fac[r] = __expf(mrow[r] - mn);
      mrow[r] = mn;
      float p0 = __expf(s0 - mn), p1 = __expf(s1 - mn);
      Ps[wid][hi * 8 + r][lm]      = (bf16_t)p0;   // key tile 0
      Ps[wid][hi * 8 + r][16 + lm] = (bf16_t)p1;   // key tile 1
      float rs = p0 + p1;
#pragma unroll
      for (int m = 1; m < 16; m <<= 1) rs += __shfl_xor(rs, m);
      lrow[r] = lrow[r] * fac[r] + rs;
    }
#pragma unroll
    for (int j = 0; j < 8; ++j)
#pragma unroll
      for (int r = 0; r < 8; ++r) o[j][r] *= fac[r];

    // ---- P as A-fragment (wave-local LDS, in-order DS) ----
    Frag pa;
    {
      const unsigned* src = reinterpret_cast<const unsigned*>(&Ps[wid][lm][0]);
#pragma unroll
      for (int j = 0; j < 8; ++j) {
        int kk = 2 * j + hi * 8 + ((j >= 4) ? 8 : 0);
        pa.u[j] = src[kk >> 1];
      }
    }
    // ---- O += P * V  (8 d-subtiles of 16) ----
#pragma unroll
    for (int t = 0; t < 8; ++t) {
      Frag vb;
      const unsigned* src =
          reinterpret_cast<const unsigned*>(&Vs[cur][t * 16 + lm][0]);
#pragma unroll
      for (int j = 0; j < 8; ++j) {
        int kk = 2 * j + hi * 8 + ((j >= 4) ? 8 : 0);
        vb.u[j] = src[kk >> 1];
      }
      o[t] = wmma_bf16(pa.v, vb.v, o[t]);
    }

    async_wait0();
    __syncthreads();
  }

  // ---- epilogue: normalize and write ctx (b,n,h,d) bf16 ----
#pragma unroll
  for (int t = 0; t < 8; ++t)
#pragma unroll
    for (int r = 0; r < 8; ++r) {
      int qrow = q0 + hi * 8 + r;
      float val = o[t][r] / lrow[r];
      ctx[((size_t)(b * SEQ + qrow) * NH + h) * HDIM + t * 16 + lm] = (bf16_t)val;
    }
}

// ---------------------------------------------------------------------------
// Host-side orchestration.
// ---------------------------------------------------------------------------
extern "C" void kernel_launch(void* const* d_in, const int* in_sizes, int n_in,
                              void* d_out, int out_size, void* d_ws, size_t ws_size,
                              hipStream_t stream) {
  const float* hs = (const float*)d_in[0];
  const float* rc = (const float*)d_in[1];
  const float* rs = (const float*)d_in[2];
  const float* Wq = (const float*)d_in[3];
  const float* Wk = (const float*)d_in[4];
  const float* Wv = (const float*)d_in[5];
  const float* Wo = (const float*)d_in[6];
  const float* qw = (const float*)d_in[7];
  const float* kw = (const float*)d_in[8];
  float* out = (float*)d_out;

  // Workspace carve-up (~120 MB total)
  char* p = (char*)d_ws;
  bf16_t* hbf = (bf16_t*)p; p += (size_t)BN * DIM * sizeof(bf16_t);    // 16 MB
  bf16_t* wbf = (bf16_t*)p; p += (size_t)DIM * DIM * sizeof(bf16_t);   //  8 MB
  float*  Pf  = (float*)p;  p += (size_t)BN * DIM * sizeof(float);     // 32 MB
  bf16_t* qb  = (bf16_t*)p; p += (size_t)BN * DIM * sizeof(bf16_t);    // 16 MB
  bf16_t* kb  = (bf16_t*)p; p += (size_t)BN * DIM * sizeof(bf16_t);    // 16 MB
  bf16_t* vb  = (bf16_t*)p; p += (size_t)BN * DIM * sizeof(bf16_t);    // 16 MB
  bf16_t* cx  = (bf16_t*)p; p += (size_t)BN * DIM * sizeof(bf16_t);    // 16 MB

  const size_t nHid = (size_t)BN * DIM;    // 8,388,608
  const size_t nW   = (size_t)DIM * DIM;   // 4,194,304
  dim3 gemmGrid(DIM / 128, BN / 128);      // (16, 32)
  dim3 attnGrid(SEQ / 128, BATCH * NH);    // (16, 32)

  cvt_f32_bf16<<<(int)(nHid / 256), 256, 0, stream>>>(hs, hbf, nHid);

  // Q path
  cvt_f32_bf16<<<(int)(nW / 256), 256, 0, stream>>>(Wq, wbf, nW);
  gemm_bf16<<<gemmGrid, 256, 0, stream>>>(hbf, wbf, Pf, BN, DIM, DIM);
  rmsnorm_rope<<<(BN * NH) / 8, 256, 0, stream>>>(Pf, qw, rc, rs, qb);

  // K path
  cvt_f32_bf16<<<(int)(nW / 256), 256, 0, stream>>>(Wk, wbf, nW);
  gemm_bf16<<<gemmGrid, 256, 0, stream>>>(hbf, wbf, Pf, BN, DIM, DIM);
  rmsnorm_rope<<<(BN * NH) / 8, 256, 0, stream>>>(Pf, kw, rc, rs, kb);

  // V path
  cvt_f32_bf16<<<(int)(nW / 256), 256, 0, stream>>>(Wv, wbf, nW);
  gemm_bf16<<<gemmGrid, 256, 0, stream>>>(hbf, wbf, Pf, BN, DIM, DIM);
  v_cvt<<<(int)(nHid / 256), 256, 0, stream>>>(Pf, vb);

  // Attention
  flash_attn<<<attnGrid, 256, 0, stream>>>(qb, kb, vb, cx);

  // Output projection
  cvt_f32_bf16<<<(int)(nW / 256), 256, 0, stream>>>(Wo, wbf, nW);
  gemm_bf16<<<gemmGrid, 256, 0, stream>>>(cx, wbf, out, BN, DIM, DIM);
}
